// InteractionBlock_80401787781523
// MI455X (gfx1250) — compile-verified, matrix operand order checked by reading, not run on hardware
//
#include <hip/hip_runtime.h>
#include <hip/hip_fp16.h>

typedef __attribute__((ext_vector_type(16))) _Float16 v16h;
typedef __attribute__((ext_vector_type(8)))  float    v8f;

#define CCH 32

__device__ __forceinline__ float silu(float x) { return x / (1.0f + __expf(-x)); }

// A/B 16-bit operand K-index for element i of the per-lane 16-vector.
// lanes 0-15 (hi=0): i0..7 -> K0..7,  i8..15 -> K16..23
// lanes 16-31(hi=1): i0..7 -> K8..15, i8..15 -> K24..31   (ISA 7.12.2)
__device__ __forceinline__ int kmap(int i, int hi) {
    return hi * 8 + (i < 8 ? i : 8 + i);
}

__device__ __forceinline__ v8f wmma_f16(v16h a, v16h b, v8f c) {
    return __builtin_amdgcn_wmma_f32_16x16x32_f16(false, a, false, b, (short)0, c, false, false);
}

// ---------------- kernel 0: zero the edge-aggregation accumulators ----------
__global__ void zero_kernel(float* __restrict__ p, size_t n) {
    size_t i = (size_t)blockIdx.x * 256 + threadIdx.x;
    if (i < n) p[i] = 0.0f;
}

// ---------------- kernel 1: node up-projection (WMMA) -----------------------
// s_up = s @ W_up_s * 1/sqrt(C)          (N,32)
// v_up[:,:,x] = v[:,:,x] @ W_up_v * 1/sqrt(C)   stored (N,C,3)
__global__ __launch_bounds__(32) void node_up_kernel(
    const float* __restrict__ nf, const float* __restrict__ Wus,
    const float* __restrict__ Wuv, float* __restrict__ s_up,
    float* __restrict__ v_up, int N)
{
    const int lane = threadIdx.x;
    const int col  = lane & 15;
    const int hi   = lane >> 4;
    const int base = blockIdx.x * 16;
    const float invC = 0.17677669529663688f; // 1/sqrt(32)

    int arow = base + col; if (arow >= N) arow = N - 1;
    const float* nfr = nf + (long long)arow * 128;

    v16h aS, aV0, aV1, aV2;
    #pragma unroll
    for (int i = 0; i < 16; ++i) {
        int k = kmap(i, hi);
        aS[i]  = (_Float16)nfr[k];
        aV0[i] = (_Float16)nfr[32 + k * 3 + 0];
        aV1[i] = (_Float16)nfr[32 + k * 3 + 1];
        aV2[i] = (_Float16)nfr[32 + k * 3 + 2];
    }

    v8f cS[2], cV[3][2];
    #pragma unroll
    for (int nt = 0; nt < 2; ++nt) {
        int d = nt * 16 + col;
        v16h bS, bV;
        #pragma unroll
        for (int i = 0; i < 16; ++i) {
            int k = kmap(i, hi);
            bS[i] = (_Float16)Wus[k * 32 + d];
            bV[i] = (_Float16)Wuv[k * 32 + d];
        }
        v8f z = {};
        cS[nt]    = wmma_f16(aS,  bS, z);
        cV[0][nt] = wmma_f16(aV0, bV, z);
        cV[1][nt] = wmma_f16(aV1, bV, z);
        cV[2][nt] = wmma_f16(aV2, bV, z);
    }

    #pragma unroll
    for (int r = 0; r < 8; ++r) {
        int m = base + r + 8 * hi;
        if (m >= N) continue;
        #pragma unroll
        for (int nt = 0; nt < 2; ++nt) {
            int d = nt * 16 + col;
            s_up[(long long)m * 32 + d] = cS[nt][r] * invC;
            #pragma unroll
            for (int x = 0; x < 3; ++x)
                v_up[(long long)m * 96 + d * 3 + x] = cV[x][nt][r] * invC;
        }
    }
}

// ---------------- kernel 2: per-edge messages + atomic scatter --------------
// one wave per edge, lane = channel c
__global__ __launch_bounds__(256) void edge_kernel(
    const float* __restrict__ ef, const float* __restrict__ radial,
    const int* __restrict__ senders, const int* __restrict__ receivers,
    const float* __restrict__ Wr1, const float* __restrict__ Wr2,
    const float* __restrict__ s_up, const float* __restrict__ v_up,
    float* __restrict__ agg0, float* __restrict__ agg1, int E)
{
    const int wid = threadIdx.x >> 5;
    const int c   = threadIdx.x & 31;
    long long e = (long long)blockIdx.x * 8 + wid;
    if (e >= E) return;

    const float invR   = 0.35355339059327373f; // 1/sqrt(8)
    const float inv3   = 0.5773502691896258f;  // 1/sqrt(3)
    const float degsc  = 0.25f;                // 1/sqrt(16)

    const float* re = radial + e * 8;
    float h[8];
    #pragma unroll
    for (int j = 0; j < 8; ++j) {
        float acc = 0.0f;
        #pragma unroll
        for (int k = 0; k < 8; ++k) acc += re[k] * Wr1[k * 8 + j];
        h[j] = silu(acc * invR);
    }
    float w0 = 0.f, w1 = 0.f, w2 = 0.f, w3 = 0.f;
    #pragma unroll
    for (int j = 0; j < 8; ++j) {
        const float* wr = Wr2 + j * 128;
        w0 += h[j] * wr[c];
        w1 += h[j] * wr[32 + c];
        w2 += h[j] * wr[64 + c];
        w3 += h[j] * wr[96 + c];
    }
    w0 *= invR; w1 *= invR; w2 *= invR; w3 *= invR;

    float sh0  = ef[e * 4 + 0];
    float sh1x = ef[e * 4 + 1];
    float sh1y = ef[e * 4 + 2];
    float sh1z = ef[e * 4 + 3];

    long long snd = senders[e];
    float se = s_up[snd * 32 + c];
    const float* vp = v_up + snd * 96 + c * 3;
    float vx = vp[0], vy = vp[1], vz = vp[2];
    float dotv = vx * sh1x + vy * sh1y + vz * sh1z;

    long long rn = receivers[e];
    float* a0 = agg0 + rn * 64;
    float* a1 = agg1 + rn * 192;

    atomicAdd(&a0[c],      w0 * se * sh0 * degsc);
    atomicAdd(&a0[32 + c], w3 * dotv * inv3 * degsc);
    float m1s = w1 * se * degsc;
    atomicAdd(&a1[c * 3 + 0], m1s * sh1x);
    atomicAdd(&a1[c * 3 + 1], m1s * sh1y);
    atomicAdd(&a1[c * 3 + 2], m1s * sh1z);
    float m1v = w2 * sh0 * degsc;
    atomicAdd(&a1[(32 + c) * 3 + 0], m1v * vx);
    atomicAdd(&a1[(32 + c) * 3 + 1], m1v * vy);
    atomicAdd(&a1[(32 + c) * 3 + 2], m1v * vz);
}

// ---------------- kernel 3: down-projection (WMMA) + skip + gate ------------
__global__ __launch_bounds__(32) void down_kernel(
    const float* __restrict__ agg0, const float* __restrict__ agg1,
    const float* __restrict__ nf, const int* __restrict__ species,
    const float* __restrict__ Wds, const float* __restrict__ Wdv,
    const float* __restrict__ Wss, const float* __restrict__ Wsv,
    float* __restrict__ out, int N)
{
    __shared__ float sS[16 * 32];
    __shared__ float sV[16 * 96];
    __shared__ int   sSpec[16];

    const int lane = threadIdx.x;
    const int col  = lane & 15;
    const int hi   = lane >> 4;
    const int base = blockIdx.x * 16;

    for (int idx = lane; idx < 16 * 128; idx += 32) {
        int rr = idx >> 7;
        int f  = idx & 127;
        int m = base + rr; if (m >= N) m = N - 1;
        float v = nf[(long long)m * 128 + f];
        if (f < 32) sS[rr * 32 + f] = v;
        else        sV[rr * 96 + (f - 32)] = v;
    }
    if (lane < 16) {
        int m = base + lane; if (m >= N) m = N - 1;
        sSpec[lane] = species[m];
    }
    __syncthreads();

    int arow = base + col; if (arow >= N) arow = N - 1;
    const float* a0r = agg0 + (long long)arow * 64;
    const float* a1r = agg1 + (long long)arow * 192;

    v16h aS[2], aV[3][2];
    #pragma unroll
    for (int kt = 0; kt < 2; ++kt) {
        #pragma unroll
        for (int i = 0; i < 16; ++i) {
            int k = kt * 32 + kmap(i, hi);
            aS[kt][i]    = (_Float16)a0r[k];
            aV[0][kt][i] = (_Float16)a1r[k * 3 + 0];
            aV[1][kt][i] = (_Float16)a1r[k * 3 + 1];
            aV[2][kt][i] = (_Float16)a1r[k * 3 + 2];
        }
    }

    v8f accS[4];
    #pragma unroll
    for (int nt = 0; nt < 4; ++nt) {
        int d = nt * 16 + col;
        v16h b0, b1;
        #pragma unroll
        for (int i = 0; i < 16; ++i) {
            int k = kmap(i, hi);
            b0[i] = (_Float16)Wds[k * 64 + d];
            b1[i] = (_Float16)Wds[(k + 32) * 64 + d];
        }
        v8f z = {};
        v8f t = wmma_f16(aS[0], b0, z);
        accS[nt] = wmma_f16(aS[1], b1, t);
    }

    v8f accV[3][2];
    #pragma unroll
    for (int x = 0; x < 3; ++x) {
        #pragma unroll
        for (int nt = 0; nt < 2; ++nt) {
            int d = nt * 16 + col;
            v16h b0, b1;
            #pragma unroll
            for (int i = 0; i < 16; ++i) {
                int k = kmap(i, hi);
                b0[i] = (_Float16)Wdv[k * 32 + d];
                b1[i] = (_Float16)Wdv[(k + 32) * 32 + d];
            }
            v8f z = {};
            v8f t = wmma_f16(aV[x][0], b0, z);
            accV[x][nt] = wmma_f16(aV[x][1], b1, t);
        }
    }

    const float inv2C = 0.125f;                 // 1/sqrt(64)
    const float invC  = 0.17677669529663688f;   // 1/sqrt(32)

    #pragma unroll
    for (int r = 0; r < 8; ++r) {
        int rr = r + 8 * hi;
        int m = base + rr;
        bool valid = (m < N);
        int spec = sSpec[rr];
        const float* ws_s = Wss + spec * 2048; // (32,64)
        const float* ws_v = Wsv + spec * 1024; // (32,32)

        float scv[4];
        #pragma unroll
        for (int nt = 0; nt < 4; ++nt) {
            int d = nt * 16 + col;
            float sk = 0.0f;
            for (int c2 = 0; c2 < 32; ++c2) sk += sS[rr * 32 + c2] * ws_s[c2 * 64 + d];
            scv[nt] = 0.5f * (accS[nt][r] * inv2C + sk * invC);
        }
        if (valid) {
            out[(long long)m * 128 + col]      = silu(scv[0]);
            out[(long long)m * 128 + 16 + col] = silu(scv[1]);
        }
        float g[2] = { silu(scv[2]), silu(scv[3]) };

        #pragma unroll
        for (int nt = 0; nt < 2; ++nt) {
            int d = nt * 16 + col; // vector channel
            #pragma unroll
            for (int x = 0; x < 3; ++x) {
                float sk = 0.0f;
                for (int c2 = 0; c2 < 32; ++c2) sk += sV[rr * 96 + c2 * 3 + x] * ws_v[c2 * 32 + d];
                float vg = 0.5f * (accV[x][nt][r] * inv2C + sk * invC) * g[nt];
                if (valid) out[(long long)m * 128 + 32 + d * 3 + x] = vg;
            }
        }
    }
}

extern "C" void kernel_launch(void* const* d_in, const int* in_sizes, int n_in,
                              void* d_out, int out_size, void* d_ws, size_t ws_size,
                              hipStream_t stream)
{
    const float* nf        = (const float*)d_in[0];
    const float* ef        = (const float*)d_in[1];
    const float* radial    = (const float*)d_in[2];
    const int*   senders   = (const int*)d_in[3];
    const int*   receivers = (const int*)d_in[4];
    const int*   species   = (const int*)d_in[5];
    const float* Wus       = (const float*)d_in[6];
    const float* Wuv       = (const float*)d_in[7];
    const float* Wr1       = (const float*)d_in[8];
    const float* Wr2       = (const float*)d_in[9];
    const float* Wds       = (const float*)d_in[10];
    const float* Wdv       = (const float*)d_in[11];
    const float* Wss       = (const float*)d_in[12];
    const float* Wsv       = (const float*)d_in[13];

    const int N = in_sizes[5];
    const int E = in_sizes[3];

    float* ws   = (float*)d_ws;
    float* s_up = ws;                       // N*32
    float* v_up = ws + (size_t)N * 32;      // N*96
    float* agg0 = ws + (size_t)N * 128;     // N*64
    float* agg1 = ws + (size_t)N * 192;     // N*192
    float* outp = (float*)d_out;

    size_t zn = (size_t)N * 256;
    zero_kernel<<<(unsigned)((zn + 255) / 256), 256, 0, stream>>>(agg0, zn);
    node_up_kernel<<<(N + 15) / 16, 32, 0, stream>>>(nf, Wus, Wuv, s_up, v_up, N);
    edge_kernel<<<(E + 7) / 8, 256, 0, stream>>>(ef, radial, senders, receivers,
                                                 Wr1, Wr2, s_up, v_up, agg0, agg1, E);
    down_kernel<<<(N + 15) / 16, 32, 0, stream>>>(agg0, agg1, nf, species,
                                                  Wds, Wdv, Wss, Wsv, outp, N);
}
